// CausalSelfAttention_73358041415963
// MI455X (gfx1250) — matrix operand order repaired and checked by reading, not compiled, not run on hardware
//
#include <hip/hip_runtime.h>
#include <hip/hip_bf16.h>

// Problem constants (match reference)
constexpr int B_  = 2;
constexpr int N_  = 2048;
constexpr int D_  = 1024;
constexpr int H_  = 16;
constexpr int DK_ = 64;
constexpr int DV_ = 64;

typedef __attribute__((ext_vector_type(16))) __bf16 v16bf;
typedef __attribute__((ext_vector_type(8)))  __bf16 v8bf;
typedef __attribute__((ext_vector_type(8)))  float  v8f;

// ---------------------------------------------------------------------------
// WMMA helpers: layouts per CDNA5 ISA 7.12.2
//   A (16x32 bf16): lane row = l&15; halves e<8 -> k = base+e, e>=8 -> k = base+16+(e-8),
//                   base = 8*(l>>4)
//   B (32x16 bf16, loaded from BT[n][k] row-major): lane col = l&15; same k map as A
//   C/D (16x16 f32): m = r + 8*(l>>4), n = l&15
// ---------------------------------------------------------------------------
__device__ __forceinline__ v8f wmma_bf16(v16bf a, v16bf b, v8f c) {
  return __builtin_amdgcn_wmma_f32_16x16x32_bf16(false, a, false, b, (short)0, c,
                                                 false, false);
}

__device__ __forceinline__ v16bf load_a_frag(const __bf16* A, int lda, int m0, int k0,
                                             int lane) {
  const __bf16* p = A + (size_t)(m0 + (lane & 15)) * lda + k0 + ((lane >> 4) << 3);
  union { v16bf v; v8bf h[2]; } u;
  u.h[0] = *(const v8bf*)(p);
  u.h[1] = *(const v8bf*)(p + 16);
  return u.v;
}

// BT is the transposed B matrix, i.e. BT[n][k] row-major with leading dim ldb.
__device__ __forceinline__ v16bf load_bt_frag(const __bf16* BT, int ldb, int n0, int k0,
                                              int lane) {
  const __bf16* p = BT + (size_t)(n0 + (lane & 15)) * ldb + k0 + ((lane >> 4) << 3);
  union { v16bf v; v8bf h[2]; } u;
  u.h[0] = *(const v8bf*)(p);
  u.h[1] = *(const v8bf*)(p + 16);
  return u.v;
}

// ---------------------------------------------------------------------------
// f32 -> bf16 conversion kernels
// ---------------------------------------------------------------------------
__global__ void cvt_kernel(const float* __restrict__ in, __bf16* __restrict__ out,
                           long n) {
  long i = (long)blockIdx.x * blockDim.x + threadIdx.x;
  long stride = (long)gridDim.x * blockDim.x;
  for (; i < n; i += stride) out[i] = (__bf16)in[i];
}

// in[h][r][c] -> out[h][c][r]
__global__ void cvt_tr_kernel(const float* __restrict__ in, __bf16* __restrict__ out,
                              int Hn, int R, int C) {
  long n = (long)Hn * R * C;
  long i = (long)blockIdx.x * blockDim.x + threadIdx.x;
  long stride = (long)gridDim.x * blockDim.x;
  for (; i < n; i += stride) {
    int c = (int)(i % C);
    long t = i / C;
    int r = (int)(t % R);
    int hh = (int)(t / R);
    out[((long)hh * C + c) * R + r] = (__bf16)in[i];
  }
}

// ---------------------------------------------------------------------------
// QKV projection: 32-row x 64-col tile per wave, ping-pong double-buffered
// K loop (step 64 = 2 x 32) so fragment "rotation" needs no register moves.
// grid = (B*N/32, H, 3), block = 32 (one wave)
// ---------------------------------------------------------------------------
__global__ __launch_bounds__(32) void qkv_kernel(
    const __bf16* __restrict__ X, const __bf16* __restrict__ Wqt,
    const __bf16* __restrict__ Wkt, const __bf16* __restrict__ Wvt,
    __bf16* __restrict__ Q, __bf16* __restrict__ K, __bf16* __restrict__ Vt) {
  const int lane = threadIdx.x;
  const int m0 = blockIdx.x * 32;          // row in flattened [B*N]
  const int h  = blockIdx.y;
  const int z  = blockIdx.z;               // 0=Q 1=K 2=V
  const int b  = m0 / N_;
  const int n0 = m0 - b * N_;
  const int bh = b * H_ + h;
  const __bf16* Wt = (z == 0 ? Wqt : (z == 1 ? Wkt : Wvt)) + (size_t)h * DK_ * D_;

  v8f acc[2][4];
  #pragma unroll
  for (int i = 0; i < 2; ++i)
    #pragma unroll
    for (int t = 0; t < 4; ++t) acc[i][t] = v8f{};

  // set A primed at k = 0
  v16bf a0A = load_a_frag(X, D_, m0, 0, lane);
  v16bf a1A = load_a_frag(X, D_, m0 + 16, 0, lane);
  v16bf bfA[4];
  #pragma unroll
  for (int t = 0; t < 4; ++t) bfA[t] = load_bt_frag(Wt, D_, t * 16, 0, lane);

  for (int k0 = 0; k0 < D_; k0 += 64) {
    const int kB = k0 + 32;                         // always valid (k0 <= D_-64)
    // load set B while consuming set A
    v16bf a0B = load_a_frag(X, D_, m0, kB, lane);
    v16bf a1B = load_a_frag(X, D_, m0 + 16, kB, lane);
    v16bf bfB[4];
    #pragma unroll
    for (int t = 0; t < 4; ++t) bfB[t] = load_bt_frag(Wt, D_, t * 16, kB, lane);
    #pragma unroll
    for (int t = 0; t < 4; ++t) {
      acc[0][t] = wmma_bf16(a0A, bfA[t], acc[0][t]);
      acc[1][t] = wmma_bf16(a1A, bfA[t], acc[1][t]);
    }
    // reload set A (next k) while consuming set B
    const int kA = (k0 + 64 < D_) ? (k0 + 64) : k0;  // clamped; last reload harmless
    a0A = load_a_frag(X, D_, m0, kA, lane);
    a1A = load_a_frag(X, D_, m0 + 16, kA, lane);
    #pragma unroll
    for (int t = 0; t < 4; ++t) bfA[t] = load_bt_frag(Wt, D_, t * 16, kA, lane);
    #pragma unroll
    for (int t = 0; t < 4; ++t) {
      acc[0][t] = wmma_bf16(a0B, bfB[t], acc[0][t]);
      acc[1][t] = wmma_bf16(a1B, bfB[t], acc[1][t]);
    }
  }

  const int col  = lane & 15;
  const int rowh = (lane >> 4) * 8;
  if (z < 2) {
    __bf16* Out = (z == 0) ? Q : K;
    #pragma unroll
    for (int i = 0; i < 2; ++i)
      #pragma unroll
      for (int t = 0; t < 4; ++t)
        #pragma unroll
        for (int r = 0; r < 8; ++r) {
          int n  = n0 + i * 16 + rowh + r;
          int dv = t * 16 + col;
          Out[((size_t)bh * N_ + n) * DK_ + dv] = (__bf16)acc[i][t][r];
        }
  } else {
    #pragma unroll
    for (int i = 0; i < 2; ++i)
      #pragma unroll
      for (int t = 0; t < 4; ++t)
        #pragma unroll
        for (int r = 0; r < 8; ++r) {
          int n  = n0 + i * 16 + rowh + r;
          int dv = t * 16 + col;
          Vt[(size_t)bh * DV_ * N_ + (size_t)dv * N_ + n] = (__bf16)acc[i][t][r];
        }
  }
}

// ---------------------------------------------------------------------------
// Column-softmax stats: for each key column j of S = Q K^T, compute
//   m_j = max_i S[i,j],  zinv_j = 1 / sum_i exp(S[i,j]-m_j)
// A = K-tile (rows=keys), B = Q^T (cols=queries) => D[key][query]; reduction over
// the query axis across the 16 lanes of each half-wave. Ping-pong buffered
// Q fragments (2 query tiles per loop iteration).
// grid = (N/16 key tiles, B*H), block = 32
// ---------------------------------------------------------------------------
__global__ __launch_bounds__(32) void stats_kernel(
    const __bf16* __restrict__ Q, const __bf16* __restrict__ K,
    float* __restrict__ mstat, float* __restrict__ zinv) {
  const int lane = threadIdx.x;
  const int kt = blockIdx.x;
  const int bh = blockIdx.y;
  const __bf16* Qh = Q + (size_t)bh * N_ * DK_;
  const __bf16* Kh = K + (size_t)bh * N_ * DK_;

  v16bf aK0 = load_a_frag(Kh, DK_, kt * 16, 0, lane);
  v16bf aK1 = load_a_frag(Kh, DK_, kt * 16, 32, lane);

  float mrun[8], zrun[8];
  #pragma unroll
  for (int r = 0; r < 8; ++r) { mrun[r] = -3.0e38f; zrun[r] = 0.0f; }

  v16bf b0A = load_bt_frag(Qh, DK_, 0, 0, lane);
  v16bf b1A = load_bt_frag(Qh, DK_, 0, 32, lane);

  const int NT = N_ / 16;  // 128
  for (int qt = 0; qt < NT; qt += 2) {
    // load set B (qt+1) while consuming set A (qt)
    v16bf b0B = load_bt_frag(Qh, DK_, (qt + 1) * 16, 0, lane);
    v16bf b1B = load_bt_frag(Qh, DK_, (qt + 1) * 16, 32, lane);
    #pragma unroll
    for (int half = 0; half < 2; ++half) {
      v8f s = v8f{};
      if (half == 0) {
        s = wmma_bf16(aK0, b0A, s);
        s = wmma_bf16(aK1, b1A, s);
        // reload set A (qt+2, clamped) while consuming set B next
        const int qA = (qt + 2 < NT) ? (qt + 2) : qt;
        b0A = load_bt_frag(Qh, DK_, qA * 16, 0, lane);
        b1A = load_bt_frag(Qh, DK_, qA * 16, 32, lane);
      } else {
        s = wmma_bf16(aK0, b0B, s);
        s = wmma_bf16(aK1, b1B, s);
      }
      #pragma unroll
      for (int r = 0; r < 8; ++r) {
        float v = s[r];
        float tm = v;
        #pragma unroll
        for (int msk = 1; msk <= 8; msk <<= 1) tm = fmaxf(tm, __shfl_xor(tm, msk));
        float nm = fmaxf(mrun[r], tm);
        float e  = __expf(v - nm);
        float ts = e;
        #pragma unroll
        for (int msk = 1; msk <= 8; msk <<= 1) ts += __shfl_xor(ts, msk);
        zrun[r] = zrun[r] * __expf(mrun[r] - nm) + ts;
        mrun[r] = nm;
      }
    }
  }
  if ((lane & 15) == 0) {
    const int half = lane >> 4;
    #pragma unroll
    for (int r = 0; r < 8; ++r) {
      int j = kt * 16 + half * 8 + r;
      mstat[(size_t)bh * N_ + j] = mrun[r];
      zinv[(size_t)bh * N_ + j]  = 1.0f / zrun[r];
    }
  }
}

// ---------------------------------------------------------------------------
// AV: recompute score tiles, apply column-normalized exp, round-trip P through
// LDS into A-fragment layout, accumulate P*V with WMMA. Ping-pong buffered
// K-tile fragments (64 keys per loop iteration, 2 half-tiles of 32).
// Output written in the natural flat [b,h,n,dv] layout.
// grid = (N/16 query tiles, B*H), block = 32
// ---------------------------------------------------------------------------
__global__ __launch_bounds__(32) void av_kernel(
    const __bf16* __restrict__ Q, const __bf16* __restrict__ K,
    const __bf16* __restrict__ Vt, const float* __restrict__ mstat,
    const float* __restrict__ zinv, __bf16* __restrict__ AVc) {
  __shared__ __bf16 ldsP[16 * 32];
  const int lane = threadIdx.x;
  const int qt = blockIdx.x;
  const int bh = blockIdx.y;
  const __bf16* Qh = Q + (size_t)bh * N_ * DK_;
  const __bf16* Kh = K + (size_t)bh * N_ * DK_;
  const __bf16* Vh = Vt + (size_t)bh * DV_ * N_;
  const float*  mh = mstat + (size_t)bh * N_;
  const float*  zh = zinv + (size_t)bh * N_;

  v16bf aQ0 = load_a_frag(Qh, DK_, qt * 16, 0, lane);
  v16bf aQ1 = load_a_frag(Qh, DK_, qt * 16, 32, lane);
  v8f accV[4] = {v8f{}, v8f{}, v8f{}, v8f{}};
  const int col  = lane & 15;
  const int rowh = (lane >> 4) * 8;

  // set A: K fragments for keys [0,32)
  v16bf bkA[4];
  bkA[0] = load_bt_frag(Kh, DK_, 0, 0, lane);
  bkA[1] = load_bt_frag(Kh, DK_, 0, 32, lane);
  bkA[2] = load_bt_frag(Kh, DK_, 16, 0, lane);
  bkA[3] = load_bt_frag(Kh, DK_, 16, 32, lane);

  for (int j0 = 0; j0 < N_; j0 += 64) {
    const int jB = j0 + 32;  // always valid (j0 <= N_-64)
    // ---- half 1: keys [j0, j0+32) using set A ----
    v8f s0 = v8f{}, s1 = v8f{};
    s0 = wmma_bf16(aQ0, bkA[0], s0);
    s0 = wmma_bf16(aQ1, bkA[1], s0);
    s1 = wmma_bf16(aQ0, bkA[2], s1);
    s1 = wmma_bf16(aQ1, bkA[3], s1);
    // load set B (keys [jB, jB+32)) + V for this half, overlapping exp/LDS work
    v16bf bkB[4];
    bkB[0] = load_bt_frag(Kh, DK_, jB, 0, lane);
    bkB[1] = load_bt_frag(Kh, DK_, jB, 32, lane);
    bkB[2] = load_bt_frag(Kh, DK_, jB + 16, 0, lane);
    bkB[3] = load_bt_frag(Kh, DK_, jB + 16, 32, lane);
    v16bf bv[4];
    #pragma unroll
    for (int t = 0; t < 4; ++t) bv[t] = load_bt_frag(Vh, N_, t * 16, j0, lane);
    {
      float m0c = mh[j0 + col],      z0c = zh[j0 + col];
      float m1c = mh[j0 + 16 + col], z1c = zh[j0 + 16 + col];
      #pragma unroll
      for (int r = 0; r < 8; ++r) {
        int m = rowh + r;
        ldsP[m * 32 + col]      = (__bf16)(__expf(s0[r] - m0c) * z0c);
        ldsP[m * 32 + 16 + col] = (__bf16)(__expf(s1[r] - m1c) * z1c);
      }
      __syncthreads();
      v16bf aP = load_a_frag(ldsP, 32, 0, 0, lane);
      __syncthreads();
      #pragma unroll
      for (int t = 0; t < 4; ++t) accV[t] = wmma_bf16(aP, bv[t], accV[t]);
    }
    // ---- half 2: keys [jB, jB+32) using set B ----
    s0 = v8f{}; s1 = v8f{};
    s0 = wmma_bf16(aQ0, bkB[0], s0);
    s0 = wmma_bf16(aQ1, bkB[1], s0);
    s1 = wmma_bf16(aQ0, bkB[2], s1);
    s1 = wmma_bf16(aQ1, bkB[3], s1);
    // reload set A (keys [j0+64, ...), clamped) + V for this half
    const int jA = (j0 + 64 < N_) ? (j0 + 64) : j0;
    bkA[0] = load_bt_frag(Kh, DK_, jA, 0, lane);
    bkA[1] = load_bt_frag(Kh, DK_, jA, 32, lane);
    bkA[2] = load_bt_frag(Kh, DK_, jA + 16, 0, lane);
    bkA[3] = load_bt_frag(Kh, DK_, jA + 16, 32, lane);
    #pragma unroll
    for (int t = 0; t < 4; ++t) bv[t] = load_bt_frag(Vh, N_, t * 16, jB, lane);
    {
      float m0c = mh[jB + col],      z0c = zh[jB + col];
      float m1c = mh[jB + 16 + col], z1c = zh[jB + 16 + col];
      #pragma unroll
      for (int r = 0; r < 8; ++r) {
        int m = rowh + r;
        ldsP[m * 32 + col]      = (__bf16)(__expf(s0[r] - m0c) * z0c);
        ldsP[m * 32 + 16 + col] = (__bf16)(__expf(s1[r] - m1c) * z1c);
      }
      __syncthreads();
      v16bf aP = load_a_frag(ldsP, 32, 0, 0, lane);
      __syncthreads();
      #pragma unroll
      for (int t = 0; t < 4; ++t) accV[t] = wmma_bf16(aP, bv[t], accV[t]);
    }
  }

  const int n0 = qt * 16;
  #pragma unroll
  for (int t = 0; t < 4; ++t)
    #pragma unroll
    for (int r = 0; r < 8; ++r) {
      int n  = n0 + rowh + r;
      int dv = t * 16 + col;
      AVc[((size_t)bh * N_ + n) * DV_ + dv] = (__bf16)accV[t][r];
    }
}

// ---------------------------------------------------------------------------
// Output projection: [B*N x H*DV] x WoT([D x H*DV], transposed) -> f32 out
// 32-row x 64-col tile per wave, ping-pong double-buffered K loop.
// grid = (B*N/32, D/64), block = 32
// ---------------------------------------------------------------------------
__global__ __launch_bounds__(32) void proj_kernel(
    const __bf16* __restrict__ AVc, const __bf16* __restrict__ WoT,
    float* __restrict__ out) {
  const int lane = threadIdx.x;
  const int m0 = blockIdx.x * 32;
  const int n0 = blockIdx.y * 64;
  const int KD = H_ * DV_;  // 1024
  v8f acc[2][4];
  #pragma unroll
  for (int i = 0; i < 2; ++i)
    #pragma unroll
    for (int t = 0; t < 4; ++t) acc[i][t] = v8f{};

  v16bf a0A = load_a_frag(AVc, KD, m0, 0, lane);
  v16bf a1A = load_a_frag(AVc, KD, m0 + 16, 0, lane);
  v16bf bfA[4];
  #pragma unroll
  for (int t = 0; t < 4; ++t) bfA[t] = load_bt_frag(WoT, KD, n0 + t * 16, 0, lane);

  for (int k0 = 0; k0 < KD; k0 += 64) {
    const int kB = k0 + 32;
    v16bf a0B = load_a_frag(AVc, KD, m0, kB, lane);
    v16bf a1B = load_a_frag(AVc, KD, m0 + 16, kB, lane);
    v16bf bfB[4];
    #pragma unroll
    for (int t = 0; t < 4; ++t) bfB[t] = load_bt_frag(WoT, KD, n0 + t * 16, kB, lane);
    #pragma unroll
    for (int t = 0; t < 4; ++t) {
      acc[0][t] = wmma_bf16(a0A, bfA[t], acc[0][t]);
      acc[1][t] = wmma_bf16(a1A, bfA[t], acc[1][t]);
    }
    const int kA = (k0 + 64 < KD) ? (k0 + 64) : k0;
    a0A = load_a_frag(AVc, KD, m0, kA, lane);
    a1A = load_a_frag(AVc, KD, m0 + 16, kA, lane);
    #pragma unroll
    for (int t = 0; t < 4; ++t) bfA[t] = load_bt_frag(WoT, KD, n0 + t * 16, kA, lane);
    #pragma unroll
    for (int t = 0; t < 4; ++t) {
      acc[0][t] = wmma_bf16(a0B, bfB[t], acc[0][t]);
      acc[1][t] = wmma_bf16(a1B, bfB[t], acc[1][t]);
    }
  }

  const int col = lane & 15, rowh = (lane >> 4) * 8;
  #pragma unroll
  for (int i = 0; i < 2; ++i)
    #pragma unroll
    for (int t = 0; t < 4; ++t)
      #pragma unroll
      for (int r = 0; r < 8; ++r)
        out[(size_t)(m0 + i * 16 + rowh + r) * D_ + n0 + t * 16 + col] = acc[i][t][r];
}

// ---------------------------------------------------------------------------
extern "C" void kernel_launch(void* const* d_in, const int* in_sizes, int n_in,
                              void* d_out, int out_size, void* d_ws, size_t ws_size,
                              hipStream_t stream) {
  const float* X  = (const float*)d_in[0];
  const float* WQ = (const float*)d_in[1];
  const float* WK = (const float*)d_in[2];
  const float* WV = (const float*)d_in[3];
  const float* WO = (const float*)d_in[4];
  float* out = (float*)d_out;

  // Workspace carve (all sizes are multiples of 256B)
  char* ws = (char*)d_ws;
  size_t off = 0;
  auto carve = [&](size_t bytes) { void* p = ws + off; off += bytes; return p; };
  __bf16* Xbf  = (__bf16*)carve((size_t)B_ * N_ * D_ * 2);        // 8 MB
  __bf16* Wqt  = (__bf16*)carve((size_t)H_ * D_ * DK_ * 2);       // 2 MB
  __bf16* Wkt  = (__bf16*)carve((size_t)H_ * D_ * DK_ * 2);       // 2 MB
  __bf16* Wvt  = (__bf16*)carve((size_t)H_ * D_ * DV_ * 2);       // 2 MB
  __bf16* Wot  = (__bf16*)carve((size_t)H_ * DV_ * D_ * 2);       // 2 MB
  __bf16* Qb   = (__bf16*)carve((size_t)B_ * H_ * N_ * DK_ * 2);  // 8 MB
  __bf16* Kb   = (__bf16*)carve((size_t)B_ * H_ * N_ * DK_ * 2);  // 8 MB
  __bf16* Vtb  = (__bf16*)carve((size_t)B_ * H_ * DV_ * N_ * 2);  // 8 MB
  float*  mst  = (float*)carve((size_t)B_ * H_ * N_ * 4);         // 256 KB
  float*  ziv  = (float*)carve((size_t)B_ * H_ * N_ * 4);         // 256 KB
  __bf16* AVc  = (__bf16*)carve((size_t)B_ * H_ * N_ * DV_ * 2);  // 8 MB

  // 1) convert inputs to bf16 (weights transposed so B-fragments load contiguously)
  cvt_kernel<<<2048, 256, 0, stream>>>(X, Xbf, (long)B_ * N_ * D_);
  cvt_tr_kernel<<<1024, 256, 0, stream>>>(WQ, Wqt, H_, D_, DK_);
  cvt_tr_kernel<<<1024, 256, 0, stream>>>(WK, Wkt, H_, D_, DK_);
  cvt_tr_kernel<<<1024, 256, 0, stream>>>(WV, Wvt, H_, D_, DV_);
  cvt_tr_kernel<<<1024, 256, 0, stream>>>(WO, Wot, 1, H_ * DV_, D_);

  // 2) Q/K/V projections (V stored transposed)
  qkv_kernel<<<dim3(B_ * N_ / 32, H_, 3), 32, 0, stream>>>(Xbf, Wqt, Wkt, Wvt,
                                                           Qb, Kb, Vtb);
  // 3) column-softmax statistics (softmax over the QUERY axis, per reference)
  stats_kernel<<<dim3(N_ / 16, B_ * H_), 32, 0, stream>>>(Qb, Kb, mst, ziv);
  // 4) AV with normalized probabilities
  av_kernel<<<dim3(N_ / 16, B_ * H_), 32, 0, stream>>>(Qb, Kb, Vtb, mst, ziv, AVc);
  // 5) output projection -> f32
  proj_kernel<<<dim3(B_ * N_ / 32, D_ / 64), 32, 0, stream>>>(AVc, Wot, out);
}
